// GL_GATConv_3l_128h_44753559224366
// MI455X (gfx1250) — matrix operand order repaired and checked by reading, not compiled, not run on hardware
//
#include <hip/hip_runtime.h>

// ---------------------------------------------------------------------------
// Problem constants (from reference)
// ---------------------------------------------------------------------------
constexpr int NN   = 50000;   // nodes
constexpr int EE   = 500000;  // edges
constexpr int F_IN = 128;
constexpr int HC   = 256;     // H*C
constexpr int EMAX = EE + NN; // edges incl. self loops
constexpr int NCLS = 10;

// ---------------------------------------------------------------------------
// WMMA types (CDNA5 / gfx1250, wave32)
// ---------------------------------------------------------------------------
typedef __attribute__((ext_vector_type(16))) __bf16 v16bf;
typedef __attribute__((ext_vector_type(8)))  float  v8f;

union FragBF {
    uint4 q[2];
    unsigned short s[16];
    v16bf v;
};

// ---------------------------------------------------------------------------
// Helpers
// ---------------------------------------------------------------------------
__device__ inline unsigned short f32_to_bf16_bits(float f) {
    unsigned int u = __float_as_uint(f);
    // round-to-nearest-even
    unsigned int r = (u + 0x7FFFu + ((u >> 16) & 1u)) >> 16;
    return (unsigned short)r;
}

__device__ inline void atomicMaxF(float* addr, float v) {
    if (v >= 0.0f) atomicMax((int*)addr, __float_as_int(v));
    else           atomicMin((unsigned int*)addr, __float_as_uint(v));
}

// ---------------------------------------------------------------------------
// f32 -> bf16 conversion
// ---------------------------------------------------------------------------
__global__ void k_f32_to_bf16(const float* __restrict__ in,
                              unsigned short* __restrict__ out, int n) {
    int i = blockIdx.x * blockDim.x + threadIdx.x;
    if (i < n) out[i] = f32_to_bf16_bits(in[i]);
}

// ---------------------------------------------------------------------------
// Deterministic mean of edge weights (two stage reduction)
// ---------------------------------------------------------------------------
__global__ void k_reduce_partial(const float* __restrict__ w,
                                 float* __restrict__ partial, int n) {
    __shared__ float sm[256];
    float s = 0.0f;
    for (int i = blockIdx.x * 256 + threadIdx.x; i < n; i += 256 * gridDim.x)
        s += w[i];
    sm[threadIdx.x] = s;
    __syncthreads();
    for (int off = 128; off > 0; off >>= 1) {
        if (threadIdx.x < off) sm[threadIdx.x] += sm[threadIdx.x + off];
        __syncthreads();
    }
    if (threadIdx.x == 0) partial[blockIdx.x] = sm[0];
}

__global__ void k_reduce_final(const float* __restrict__ partial,
                               float* __restrict__ wmean, float invN) {
    __shared__ float sm[256];
    sm[threadIdx.x] = partial[threadIdx.x];
    __syncthreads();
    for (int off = 128; off > 0; off >>= 1) {
        if (threadIdx.x < off) sm[threadIdx.x] += sm[threadIdx.x + off];
        __syncthreads();
    }
    if (threadIdx.x == 0) wmean[0] = sm[0] * invN;
}

// ---------------------------------------------------------------------------
// WMMA GEMM:  C[M,256] = A[M,K](bf16) * B[K,256](bf16) + bias
// Block = 512 threads = 16 waves; wave w owns 16x16 tile at cols [16w,16w+16).
// Block b owns rows [16b, 16b+16).  M == 50000 == 3125*16 exactly.
// Fragment layouts per CDNA5 ISA 7.12.2 (bf16 A 16x32, B 32x16, C f32 16x16).
// ---------------------------------------------------------------------------
__global__ __launch_bounds__(512)
void k_gemm_bf16(const unsigned short* __restrict__ A,   // [M,K] bf16 bits
                 const unsigned short* __restrict__ B,   // [K,256] bf16 bits
                 const float* __restrict__ bias,         // [256]
                 float* __restrict__ C,                  // [M,256] f32
                 int K) {
    const int lane = threadIdx.x & 31;
    const int wave = threadIdx.x >> 5;       // 0..15
    const int row0 = blockIdx.x * 16;
    const int n0   = wave * 16;
    const int g    = lane >> 4;              // half-wave select
    const int mh   = lane & 15;              // row (A) / col (B,C) within tile

    v8f acc = {0.f, 0.f, 0.f, 0.f, 0.f, 0.f, 0.f, 0.f};

    const unsigned short* Arow = A + (size_t)(row0 + mh) * K;

    for (int k0 = 0; k0 < K; k0 += 32) {
        FragBF a, b;
        // A lane(16g+m'): elems 0..7 -> k = k0+8g+e ; elems 8..15 -> k = k0+8g+16+(e-8)
        a.q[0] = *(const uint4*)(Arow + k0 + 8 * g);
        a.q[1] = *(const uint4*)(Arow + k0 + 8 * g + 16);
        // B lane(16g+n'): col n0+n'; elems 0..7 -> k = k0+8g+e ; elems 8..15 -> k = k0+8g+16+(e-8)
        const unsigned short* Bcol = B + (size_t)(k0 + 8 * g) * HC + n0 + mh;
        for (int e = 0; e < 8; ++e) {
            b.s[e]     = Bcol[(size_t)e * HC];
            b.s[8 + e] = Bcol[(size_t)(16 + e) * HC];
        }
        acc = __builtin_amdgcn_wmma_f32_16x16x32_bf16(
            false, a.v, false, b.v, (short)0, acc, false, false);
    }

    // C lane(16g+n'): vgpr v holds M = v + 8g, N = n'
#pragma unroll
    for (int v = 0; v < 8; ++v) {
        int r = row0 + v + 8 * g;
        C[(size_t)r * HC + n0 + mh] = acc[v] + bias[n0 + mh];
    }
}

// ---------------------------------------------------------------------------
// Per-layer init: zero accumulator, amax = -inf, denom = 0
// ---------------------------------------------------------------------------
__global__ void k_init_layer(float* __restrict__ outAcc,
                             float* __restrict__ amax,
                             float* __restrict__ denom, int n) {
    int i = blockIdx.x * blockDim.x + threadIdx.x;
    if (i < n) outAcc[i] = 0.0f;
    if (i < NN * 2) {
        amax[i]  = __uint_as_float(0xFF800000u);  // -inf
        denom[i] = 0.0f;
    }
}

// ---------------------------------------------------------------------------
// Edge pass 1: message + leaky_relu + attention dot + segment max
// One wave32 per edge; lane handles 8 channels; half-wave = one head.
// ---------------------------------------------------------------------------
__global__ __launch_bounds__(256)
void k_edge_logits(const int* __restrict__ src, const int* __restrict__ dst,
                   const float* __restrict__ ew, const float* __restrict__ wmean,
                   const float* __restrict__ xl, const float* __restrict__ xr,
                   const float* __restrict__ We, const float* __restrict__ att,
                   float* __restrict__ logits, float* __restrict__ amax,
                   int E0, int Etot) {
    int e = blockIdx.x * 8 + (threadIdx.x >> 5);
    if (e >= Etot) return;
    int lane = threadIdx.x & 31;

    int s, d; float w;
    if (e < E0) { s = src[e]; d = dst[e]; w = ew[e]; }
    else        { s = e - E0; d = s;      w = wmean[0]; }

    int c0 = lane * 8;
    const float4* pl = (const float4*)(xl + (size_t)s * HC + c0);
    const float4* pr = (const float4*)(xr + (size_t)d * HC + c0);
    const float4* pw = (const float4*)(We + c0);
    const float4* pa = (const float4*)(att + c0);

    float p = 0.0f;
#pragma unroll
    for (int i = 0; i < 2; ++i) {
        float4 a = pl[i], b = pr[i], we = pw[i], at = pa[i];
        float m;
        m = a.x + b.x + w * we.x; m = (m > 0.f) ? m : 0.2f * m; p += m * at.x;
        m = a.y + b.y + w * we.y; m = (m > 0.f) ? m : 0.2f * m; p += m * at.y;
        m = a.z + b.z + w * we.z; m = (m > 0.f) ? m : 0.2f * m; p += m * at.z;
        m = a.w + b.w + w * we.w; m = (m > 0.f) ? m : 0.2f * m; p += m * at.w;
    }
    // reduce over 16 lanes (one head per half-wave)
    p += __shfl_down(p, 8, 16);
    p += __shfl_down(p, 4, 16);
    p += __shfl_down(p, 2, 16);
    p += __shfl_down(p, 1, 16);

    if ((lane & 15) == 0) {
        int h = lane >> 4;
        logits[(size_t)e * 2 + h] = p;
        atomicMaxF(&amax[(size_t)d * 2 + h], p);
    }
}

// ---------------------------------------------------------------------------
// Edge pass 2: exp(logit - amax[dst]) + segment sum
// ---------------------------------------------------------------------------
__global__ void k_edge_exp(const int* __restrict__ dst,
                           const float* __restrict__ logits,
                           const float* __restrict__ amax,
                           float* __restrict__ alphaBuf,
                           float* __restrict__ denom, int E0, int Etot) {
    int idx = blockIdx.x * blockDim.x + threadIdx.x;
    if (idx >= Etot * 2) return;
    int e = idx >> 1, h = idx & 1;
    int d = (e < E0) ? dst[e] : (e - E0);
    float ex = __expf(logits[idx] - amax[(size_t)d * 2 + h]);
    alphaBuf[idx] = ex;
    atomicAdd(&denom[(size_t)d * 2 + h], ex);
}

// ---------------------------------------------------------------------------
// Edge pass 3: scatter  out[dst] += alpha * xl[src]
// ---------------------------------------------------------------------------
__global__ __launch_bounds__(256)
void k_edge_aggregate(const int* __restrict__ src, const int* __restrict__ dst,
                      const float* __restrict__ xl,
                      const float* __restrict__ alphaBuf,
                      const float* __restrict__ denom,
                      float* __restrict__ outAcc, int E0, int Etot) {
    int e = blockIdx.x * 8 + (threadIdx.x >> 5);
    if (e >= Etot) return;
    int lane = threadIdx.x & 31;

    int s, d;
    if (e < E0) { s = src[e]; d = dst[e]; }
    else        { s = e - E0; d = s; }

    int h = lane >> 4;
    float coef = alphaBuf[(size_t)e * 2 + h] /
                 (denom[(size_t)d * 2 + h] + 1e-16f);

    int c0 = lane * 8;
    const float* ps = xl + (size_t)s * HC + c0;
    float* po = outAcc + (size_t)d * HC + c0;
#pragma unroll
    for (int i = 0; i < 8; ++i) atomicAdd(po + i, ps[i] * coef);
}

// ---------------------------------------------------------------------------
// Bias + ELU (in place)
// ---------------------------------------------------------------------------
__global__ void k_bias_elu(float* __restrict__ h, const float* __restrict__ bias,
                           int total) {
    int i = blockIdx.x * blockDim.x + threadIdx.x;
    if (i >= total) return;
    float v = h[i] + bias[i & (HC - 1)];
    h[i] = (v > 0.0f) ? v : (__expf(v) - 1.0f);
}

// ---------------------------------------------------------------------------
// Final classifier: out[n,k] = h[n,:] . fcW[:,k] + fcb[k]
// ---------------------------------------------------------------------------
__global__ void k_fc(const float* __restrict__ h, const float* __restrict__ W,
                     const float* __restrict__ b, float* __restrict__ out,
                     int total) {
    int idx = blockIdx.x * blockDim.x + threadIdx.x;
    if (idx >= total) return;
    int n = idx / NCLS, k = idx - n * NCLS;
    const float* hr = h + (size_t)n * HC;
    float acc = b[k];
#pragma unroll 4
    for (int c = 0; c < HC; ++c) acc += hr[c] * W[c * NCLS + k];
    out[idx] = acc;
}

// ---------------------------------------------------------------------------
// Host-side orchestration
// ---------------------------------------------------------------------------
extern "C" void kernel_launch(void* const* d_in, const int* in_sizes, int n_in,
                              void* d_out, int out_size, void* d_ws, size_t ws_size,
                              hipStream_t stream) {
    const float* x  = (const float*)d_in[0];
    const int*   ei = (const int*)d_in[1];
    const float* ew = (const float*)d_in[2];
    const int* src = ei;
    const int* dst = ei + EE;

    const float *Wl[3], *bl[3], *Wr[3], *br[3], *We[3], *att[3], *bb[3];
    for (int i = 0; i < 3; ++i) {
        int base = 3 + i * 7;
        Wl[i]  = (const float*)d_in[base + 0];
        bl[i]  = (const float*)d_in[base + 1];
        Wr[i]  = (const float*)d_in[base + 2];
        br[i]  = (const float*)d_in[base + 3];
        We[i]  = (const float*)d_in[base + 4];
        att[i] = (const float*)d_in[base + 5];
        bb[i]  = (const float*)d_in[base + 6];
    }
    const float* fcW = (const float*)d_in[24];
    const float* fcb = (const float*)d_in[25];
    float* out = (float*)d_out;

    // workspace carving (256B aligned)
    char* ws = (char*)d_ws;
    size_t off = 0;
    auto carve = [&](size_t bytes) {
        size_t r = off;
        off += (bytes + 255) & ~(size_t)255;
        return (void*)(ws + r);
    };
    float*          hbuf   = (float*)carve((size_t)NN * HC * 4);   // accum / activations
    float*          xl     = (float*)carve((size_t)NN * HC * 4);
    float*          xr     = (float*)carve((size_t)NN * HC * 4);
    unsigned short* hbf    = (unsigned short*)carve((size_t)NN * HC * 2);
    unsigned short* wbfL   = (unsigned short*)carve((size_t)HC * HC * 2);
    unsigned short* wbfR   = (unsigned short*)carve((size_t)HC * HC * 2);
    float*          logits = (float*)carve((size_t)EMAX * 2 * 4);
    float*          alphaB = (float*)carve((size_t)EMAX * 2 * 4);
    float*          amax   = (float*)carve((size_t)NN * 2 * 4);
    float*          denom  = (float*)carve((size_t)NN * 2 * 4);
    float*          parti  = (float*)carve(256 * 4);
    float*          wmean  = (float*)carve(4);
    (void)ws_size; (void)n_in; (void)in_sizes; (void)out_size;

    // self-loop edge attr = mean(weight)  (deterministic two-stage reduction)
    k_reduce_partial<<<256, 256, 0, stream>>>(ew, parti, EE);
    k_reduce_final<<<1, 256, 0, stream>>>(parti, wmean, 1.0f / (float)EE);

    const float* hin = x;
    for (int layer = 0; layer < 3; ++layer) {
        const int K    = (layer == 0) ? F_IN : HC;
        const int Etot = (layer == 0) ? EE : (EE + NN);  // layers 1,2 add self loops
        const int nh   = NN * K;
        const int nw   = K * HC;

        k_f32_to_bf16<<<(nh + 255) / 256, 256, 0, stream>>>(hin, hbf, nh);
        k_f32_to_bf16<<<(nw + 255) / 256, 256, 0, stream>>>(Wl[layer], wbfL, nw);
        k_f32_to_bf16<<<(nw + 255) / 256, 256, 0, stream>>>(Wr[layer], wbfR, nw);

        k_gemm_bf16<<<NN / 16, 512, 0, stream>>>(hbf, wbfL, bl[layer], xl, K);
        k_gemm_bf16<<<NN / 16, 512, 0, stream>>>(hbf, wbfR, br[layer], xr, K);

        k_init_layer<<<(NN * HC + 255) / 256, 256, 0, stream>>>(hbuf, amax, denom, NN * HC);

        k_edge_logits<<<(Etot + 7) / 8, 256, 0, stream>>>(
            src, dst, ew, wmean, xl, xr, We[layer], att[layer], logits, amax, EE, Etot);
        k_edge_exp<<<(Etot * 2 + 255) / 256, 256, 0, stream>>>(
            dst, logits, amax, alphaB, denom, EE, Etot);
        k_edge_aggregate<<<(Etot + 7) / 8, 256, 0, stream>>>(
            src, dst, xl, alphaB, denom, hbuf, EE, Etot);

        k_bias_elu<<<(NN * HC + 255) / 256, 256, 0, stream>>>(hbuf, bb[layer], NN * HC);
        hin = hbuf;
    }

    k_fc<<<(NN * NCLS + 255) / 256, 256, 0, stream>>>(hbuf, fcW, fcb, out, NN * NCLS);
}
